// MAPPNet_5763846111877
// MI455X (gfx1250) — matrix operand order repaired and checked by reading, not compiled, not run on hardware
//
#include <hip/hip_runtime.h>
#include <hip/hip_bf16.h>
#include <math.h>

// ---------------------------------------------------------------------------
// MAPPNet (PointNet++ MSG + LSTM) for MI455X / gfx1250.
// All GEMM-shaped work (shared MLPs, LSTM gates) runs through
// v_wmma_f32_16x16x32_f16 (f16 A/B, f32 accumulate).
// GEMM: main K-loop has no bounds clamps or masks at all (M/N overflow only
// touches never-stored rows/columns; addresses are clamped once outside the
// loop). Only the final partial-K iteration masks, and only in K.
// ---------------------------------------------------------------------------

typedef __attribute__((ext_vector_type(16))) _Float16 v16h;
typedef __attribute__((ext_vector_type(8)))  float    v8f;

constexpr int BB     = 16;    // batch
constexpr int NPTS   = 1024;  // points
constexpr int NCLS   = 13;
constexpr int NFRAME = 8;
constexpr int HDIM   = 1024;

// ---------------------------------------------------------------------------
// Generic WMMA GEMM:  Y[g,m,s] = bias[m] + sum_k W[m,k] * X[g,k,s]
// W: row-major MxK (f32, converted to f16 inline).
// X: addressed as X[g*sB + k*sK + s*sS]  (covers (B,C,S) convs and (B,H) lstm)
// Y: (G, M, S) row-major f32.
// One wave computes one 16x16 tile; 4 waves per block on consecutive N tiles.
// ---------------------------------------------------------------------------
__global__ void gemm_wmma_kernel(const float* __restrict__ W,
                                 const float* __restrict__ X,
                                 float* __restrict__ Y,
                                 const float* __restrict__ bias,
                                 int M, int K, int S, int G,
                                 long sB, long sK, long sS)
{
    const int lane  = threadIdx.x & 31;
    const int wave  = threadIdx.x >> 5;
    const int nTile = blockIdx.x * 4 + wave;
    const int mTile = blockIdx.y;
    const int g     = blockIdx.z;
    if (nTile * 16 >= S) return;            // wave-uniform exit

    const int  half  = lane >> 4;
    const int  mRow  = mTile * 16 + (lane & 15);
    const int  mCl   = mRow < M ? mRow : (M - 1);   // clamp; no masking needed
    const int  nBase = nTile * 16;
    const bool vecB  = (sK == 1);  // wave-uniform

    // A fragment layout (16-bit A 16x32): lanes 0-15 -> M; element e -> K:
    //   e in [0,8)  : ka = k0 + half*8 + e
    //   e in [8,16) : ka = k0 + 16 + half*8 + (e-8)
    const float* pArow = W + (long)mCl * K;
    const float* pA    = pArow + half * 8;

    // B fragment layout (16-bit B 32x16): lanes 0-15 -> N = nBase+(lane&15); element e -> K as for A.
    const int nCol = nBase + (lane & 15);
    const float* pB = X + (long)g * sB + (long)(nCol < S ? nCol : S - 1) * sS;   // + k*sK per step
    long offB[16];
#pragma unroll
    for (int e = 0; e < 16; ++e) {
        int kk = e < 8 ? half * 8 + e : 16 + half * 8 + (e - 8);
        offB[e] = (long)kk * sK;
    }

    v8f acc = {0.f, 0.f, 0.f, 0.f, 0.f, 0.f, 0.f, 0.f};

    // ---- main loop: full 32-wide K steps, zero clamps / zero selects ----
    const int kMain = K & ~31;
    for (int k0 = 0; k0 < kMain; k0 += 32) {
        float fa[16], fb[16];
#pragma unroll
        for (int e = 0; e < 8; ++e) {
            fa[e]     = pA[e];
            fa[e + 8] = pA[16 + e];
        }
        if (vecB) {
            const float4* p4a = (const float4*)(pB + half * 8);
            const float4* p4b = (const float4*)(pB + 16 + half * 8);
#pragma unroll
            for (int q = 0; q < 2; ++q) {
                float4 v = p4a[q];
                fb[q * 4 + 0] = v.x; fb[q * 4 + 1] = v.y;
                fb[q * 4 + 2] = v.z; fb[q * 4 + 3] = v.w;
                float4 w = p4b[q];
                fb[8 + q * 4 + 0] = w.x; fb[8 + q * 4 + 1] = w.y;
                fb[8 + q * 4 + 2] = w.z; fb[8 + q * 4 + 3] = w.w;
            }
        } else {
#pragma unroll
            for (int e = 0; e < 16; ++e) fb[e] = pB[offB[e]];
        }
        v16h a, b;
#pragma unroll
        for (int e = 0; e < 16; ++e) {
            a[e] = (_Float16)fa[e];
            b[e] = (_Float16)fb[e];
        }
        acc = __builtin_amdgcn_wmma_f32_16x16x32_f16(false, a, false, b,
                                                     (short)0, acc, false, false);
        pA += 32;
        pB += 32 * sK;
    }

    // ---- tail: one partial-K step, mask K only ----
    if (kMain < K) {
        const int  k0   = kMain;

        float fa[16], fb[16];
#pragma unroll
        for (int e = 0; e < 8; ++e) {
            int ka0 = k0 + half * 8 + e;
            int ka1 = k0 + 16 + half * 8 + e;
            fa[e]     = pArow[ka0 < K ? ka0 : K - 1];
            fa[e + 8] = pArow[ka1 < K ? ka1 : K - 1];
        }
#pragma unroll
        for (int e = 0; e < 16; ++e) {
            int kk = k0 + (e < 8 ? half * 8 + e : 16 + half * 8 + (e - 8));
            fb[e] = pB[(long)((kk < K ? kk : K - 1) - k0) * sK];
        }
        v16h a, b;
#pragma unroll
        for (int e = 0; e < 8; ++e) {
            a[e]     = (_Float16)((k0 + half * 8 + e)      < K ? fa[e]     : 0.f);
            a[e + 8] = (_Float16)((k0 + 16 + half * 8 + e) < K ? fa[e + 8] : 0.f);
        }
#pragma unroll
        for (int e = 0; e < 16; ++e) {
            int kk = k0 + (e < 8 ? half * 8 + e : 16 + half * 8 + (e - 8));
            b[e] = (_Float16)(kk < K ? fb[e] : 0.f);
        }
        acc = __builtin_amdgcn_wmma_f32_16x16x32_f16(false, a, false, b,
                                                     (short)0, acc, false, false);
    }

#pragma unroll
    for (int r = 0; r < 8; ++r) {
        int mm = mTile * 16 + r + 8 * half; // C/D layout: VGPR r -> M, lane -> N
        int ss = nBase + (lane & 15);
        if (mm < M && ss < S) {
            float v = acc[r];
            if (bias) v += bias[mm];
            Y[((long)g * M + mm) * (long)S + ss] = v;
        }
    }
}

// ---------------------------------------------------------------------------
// BatchNorm with batch statistics (reference computes mean/var inline).
// ---------------------------------------------------------------------------
__global__ void bn_stats_kernel(const float* __restrict__ Y,
                                float* __restrict__ mean, float* __restrict__ inv,
                                int M, int S, int G, float eps)
{
    const int m = blockIdx.x;
    const long NS = (long)G * S;
    float s = 0.f, sq = 0.f;
    for (long i = threadIdx.x; i < NS; i += blockDim.x) {
        long g = i / S, ss = i % S;
        float v = Y[(g * M + m) * (long)S + ss];
        s += v; sq += v * v;
    }
    __shared__ float sh[256], shq[256];
    sh[threadIdx.x] = s; shq[threadIdx.x] = sq;
    __syncthreads();
    for (int o = 128; o > 0; o >>= 1) {
        if ((int)threadIdx.x < o) {
            sh[threadIdx.x]  += sh[threadIdx.x + o];
            shq[threadIdx.x] += shq[threadIdx.x + o];
        }
        __syncthreads();
    }
    if (threadIdx.x == 0) {
        float mu  = sh[0] / (float)NS;
        float var = shq[0] / (float)NS - mu * mu;
        mean[m] = mu;
        inv[m]  = rsqrtf(var + eps);
    }
}

__global__ void bn_relu_kernel(float* __restrict__ Y,
                               const float* __restrict__ mean,
                               const float* __restrict__ inv,
                               const float* __restrict__ gam,
                               const float* __restrict__ bet,
                               int M, int S, long total)
{
    long t = (long)blockIdx.x * blockDim.x + threadIdx.x;
    if (t >= total) return;
    long ms = (long)M * S;
    long r  = t % ms;
    int  m  = (int)(r / S);
    float v = (Y[t] - mean[m]) * inv[m] * gam[m] + bet[m];
    Y[t] = v > 0.f ? v : 0.f;
}

// ---------------------------------------------------------------------------
// Pointcloud prep / grouping / sampling
// ---------------------------------------------------------------------------
__global__ void prep_kernel(const float* __restrict__ pc,
                            float* __restrict__ xyz, float* __restrict__ feats,
                            int Cin)
{
    int t = blockIdx.x * blockDim.x + threadIdx.x;
    if (t >= BB * NPTS) return;
    int b = t / NPTS, n = t % NPTS;
    const float* p = pc + (long)t * (3 + Cin);
    for (int k = 0; k < 3; ++k) xyz[(long)t * 3 + k] = p[k];
    for (int c = 0; c < Cin; ++c)
        feats[((long)b * Cin + c) * NPTS + n] = p[3 + c];
}

// Farthest-point sampling: one workgroup per batch, dist[] in LDS.
__global__ void fps_kernel(const float* __restrict__ xyz, int Npts, int npoint,
                           float* __restrict__ newXyz, int* __restrict__ outIdx)
{
    __shared__ float dist[1024];
    __shared__ float rv[256];
    __shared__ int   ri[256];
    __shared__ float lastPt[3];
    const int b = blockIdx.x, t = threadIdx.x;
    for (int j = t; j < Npts; j += blockDim.x) dist[j] = 1e10f;
    if (t == 0) {
        outIdx[b * npoint] = 0;
        for (int k = 0; k < 3; ++k) {
            float v = xyz[(long)b * Npts * 3 + k];
            newXyz[(long)b * npoint * 3 + k] = v;
            lastPt[k] = v;
        }
    }
    __syncthreads();
    for (int it = 1; it < npoint; ++it) {
        float lx = lastPt[0], ly = lastPt[1], lz = lastPt[2];
        float bv = -1.f; int bi = 0;
        for (int j = t; j < Npts; j += blockDim.x) {
            const float* p = xyz + ((long)b * Npts + j) * 3;
            float dx = p[0] - lx, dy = p[1] - ly, dz = p[2] - lz;
            float d = dx * dx + dy * dy + dz * dz;
            float nd = fminf(dist[j], d);
            dist[j] = nd;
            if (nd > bv) { bv = nd; bi = j; }
        }
        rv[t] = bv; ri[t] = bi;
        __syncthreads();
        for (int o = 128; o > 0; o >>= 1) {
            if (t < o) {
                if (rv[t + o] > rv[t] || (rv[t + o] == rv[t] && ri[t + o] < ri[t])) {
                    rv[t] = rv[t + o]; ri[t] = ri[t + o];
                }
            }
            __syncthreads();
        }
        if (t == 0) {
            int j = ri[0];
            outIdx[b * npoint + it] = j;
            for (int k = 0; k < 3; ++k) {
                float v = xyz[((long)b * Npts + j) * 3 + k];
                newXyz[((long)b * npoint + it) * 3 + k] = v;
                lastPt[k] = v;
            }
        }
        __syncthreads();
    }
}

// Ball query: first ns in-radius indices (ascending j), padded with first hit.
__global__ void ball_query_kernel(const float* __restrict__ xyz,
                                  const float* __restrict__ nx,
                                  int Npts, int P, float r2, int ns,
                                  int* __restrict__ idx)
{
    int t = blockIdx.x * blockDim.x + threadIdx.x;
    if (t >= BB * P) return;
    int b = t / P, i = t % P;
    const float* q = nx + ((long)b * P + i) * 3;
    float qx = q[0], qy = q[1], qz = q[2];
    int cnt = 0, first = -1;
    int* out = idx + (long)t * ns;
    for (int j = 0; j < Npts; ++j) {
        const float* p = xyz + ((long)b * Npts + j) * 3;
        float dx = p[0] - qx, dy = p[1] - qy, dz = p[2] - qz;
        float d = dx * dx + dy * dy + dz * dz;
        if (d < r2) {
            if (first < 0) first = j;
            if (cnt < ns) out[cnt] = j;
            ++cnt;
            if (cnt >= ns) break;
        }
    }
    int pad = first >= 0 ? first : 0;
    for (int s = cnt; s < ns; ++s) out[s] = pad;
}

// Build grouped tensor g(B, 3+Cf, P*ns): centered xyz ++ gathered feats.
__global__ void group_kernel(const float* __restrict__ xyz,
                             const float* __restrict__ nx,
                             const float* __restrict__ feats,
                             const int* __restrict__ idx,
                             int Npts, int P, int ns, int Cf,
                             float* __restrict__ g)
{
    const int Cg = 3 + Cf;
    long total = (long)BB * Cg * P * ns;
    long t = (long)blockIdx.x * blockDim.x + threadIdx.x;
    if (t >= total) return;
    int  s = (int)(t % ns); long r = t / ns;
    int  i = (int)(r % P);  r /= P;
    int  c = (int)(r % Cg);
    int  b = (int)(r / Cg);
    int  j = idx[((long)b * P + i) * ns + s];
    float v;
    if (c < 3)
        v = xyz[((long)b * Npts + j) * 3 + c] - nx[((long)b * P + i) * 3 + c];
    else
        v = feats[((long)b * Cf + (c - 3)) * Npts + j];
    g[(((long)b * Cg + c) * P + i) * ns + s] = v;
}

// sa_all grouping: g(B, 3+Cf, Npts) = xyz^T ++ feats.
__global__ void group_all_kernel(const float* __restrict__ xyz,
                                 const float* __restrict__ feats,
                                 int Npts, int Cf, float* __restrict__ g)
{
    const int Cg = 3 + Cf;
    long total = (long)BB * Cg * Npts;
    long t = (long)blockIdx.x * blockDim.x + threadIdx.x;
    if (t >= total) return;
    int n = (int)(t % Npts); long r = t / Npts;
    int c = (int)(r % Cg);
    int b = (int)(r / Cg);
    float v = (c < 3) ? xyz[((long)b * Npts + n) * 3 + c]
                      : feats[((long)b * Cf + (c - 3)) * Npts + n];
    g[((long)b * Cg + c) * Npts + n] = v;
}

// Max over nsample, write into channel slice [cOff, cOff+C) of (B, Ctot, P).
__global__ void maxpool_kernel(const float* __restrict__ Y,
                               float* __restrict__ out,
                               int C, int P, int ns, int Ctot, int cOff)
{
    long total = (long)BB * C * P;
    long t = (long)blockIdx.x * blockDim.x + threadIdx.x;
    if (t >= total) return;
    int i = (int)(t % P); long r = t / P;
    int c = (int)(r % C);
    int b = (int)(r / C);
    const float* src = Y + (((long)b * C + c) * P + i) * ns;
    float m = -1e30f;
    for (int s = 0; s < ns; ++s) m = fmaxf(m, src[s]);
    out[((long)b * Ctot + cOff + c) * P + i] = m;
}

// ---------------------------------------------------------------------------
// Feature propagation: three-NN + weighted interpolation + skip concat
// ---------------------------------------------------------------------------
__global__ void three_nn_kernel(const float* __restrict__ unknown,
                                const float* __restrict__ known,
                                int Nu, int Nk,
                                int* __restrict__ idx, float* __restrict__ wgt)
{
    int t = blockIdx.x * blockDim.x + threadIdx.x;
    if (t >= BB * Nu) return;
    int b = t / Nu, n = t % Nu;
    const float* u = unknown + ((long)b * Nu + n) * 3;
    float ux = u[0], uy = u[1], uz = u[2];
    float d0 = 1e30f, d1 = 1e30f, d2 = 1e30f;
    int i0 = 0, i1 = 0, i2 = 0;
    for (int j = 0; j < Nk; ++j) {
        const float* p = known + ((long)b * Nk + j) * 3;
        float dx = p[0] - ux, dy = p[1] - uy, dz = p[2] - uz;
        float d = dx * dx + dy * dy + dz * dz;
        if (d < d0)      { d2 = d1; i2 = i1; d1 = d0; i1 = i0; d0 = d; i0 = j; }
        else if (d < d1) { d2 = d1; i2 = i1; d1 = d; i1 = j; }
        else if (d < d2) { d2 = d; i2 = j; }
    }
    int k = Nk < 3 ? Nk : 3;
    float w0 = 1.f / (d0 + 1e-8f);
    float w1 = (k > 1) ? 1.f / (d1 + 1e-8f) : 0.f;
    float w2 = (k > 2) ? 1.f / (d2 + 1e-8f) : 0.f;
    float s = w0 + w1 + w2;
    long o = (long)t * 3;
    idx[o + 0] = i0; idx[o + 1] = (k > 1) ? i1 : 0; idx[o + 2] = (k > 2) ? i2 : 0;
    wgt[o + 0] = w0 / s; wgt[o + 1] = w1 / s; wgt[o + 2] = w2 / s;
}

// Level-3 "NN" to the single global point: idx=0, w=(1,0,0).
__global__ void nn_const_kernel(int* __restrict__ idx, float* __restrict__ wgt, int n)
{
    int t = blockIdx.x * blockDim.x + threadIdx.x;
    if (t >= n) return;
    long o = (long)t * 3;
    idx[o] = 0; idx[o + 1] = 0; idx[o + 2] = 0;
    wgt[o] = 1.f; wgt[o + 1] = 0.f; wgt[o + 2] = 0.f;
}

// x(B, Ck+Cs, Nu) = [weighted-interp(known), skip]
__global__ void interp_concat_kernel(const float* __restrict__ known,
                                     const int* __restrict__ idx,
                                     const float* __restrict__ wgt,
                                     const float* __restrict__ skip,
                                     int Ck, int Cs, int Nu, int Nk,
                                     float* __restrict__ out)
{
    const int C = Ck + Cs;
    long total = (long)BB * C * Nu;
    long t = (long)blockIdx.x * blockDim.x + threadIdx.x;
    if (t >= total) return;
    int n = (int)(t % Nu); long r = t / Nu;
    int c = (int)(r % C);
    int b = (int)(r / C);
    float v;
    if (c < Ck) {
        v = 0.f;
        long o = ((long)b * Nu + n) * 3;
        for (int k = 0; k < 3; ++k) {
            int j = idx[o + k];
            v += known[((long)b * Ck + c) * Nk + j] * wgt[o + k];
        }
    } else {
        v = skip[((long)b * Cs + (c - Ck)) * Nu + n];
    }
    out[((long)b * C + c) * Nu + n] = v;
}

// ---------------------------------------------------------------------------
// LSTM cell + classifier head
// ---------------------------------------------------------------------------
__device__ __forceinline__ float sigm(float x) { return 1.f / (1.f + expf(-x)); }

// gates laid out (4H, B): Y[m*B + n]; g1 = x@Wih^T + bih, g2 = h@Whh^T + bhh
__global__ void lstm_cell_kernel(const float* __restrict__ g1,
                                 const float* __restrict__ g2,
                                 float* __restrict__ h, float* __restrict__ c)
{
    int t = blockIdx.x * blockDim.x + threadIdx.x;
    if (t >= BB * HDIM) return;
    int b = t / HDIM, j = t % HDIM;
    float gi = g1[(long)(0 * HDIM + j) * BB + b] + g2[(long)(0 * HDIM + j) * BB + b];
    float gf = g1[(long)(1 * HDIM + j) * BB + b] + g2[(long)(1 * HDIM + j) * BB + b];
    float gg = g1[(long)(2 * HDIM + j) * BB + b] + g2[(long)(2 * HDIM + j) * BB + b];
    float go = g1[(long)(3 * HDIM + j) * BB + b] + g2[(long)(3 * HDIM + j) * BB + b];
    float cn = sigm(gf) * c[t] + sigm(gi) * tanhf(gg);
    c[t] = cn;
    h[t] = sigm(go) * tanhf(cn);
}

__global__ void copy_kernel(const float* __restrict__ src, float* __restrict__ dst, long n)
{
    long t = (long)blockIdx.x * blockDim.x + threadIdx.x;
    if (t < n) dst[t] = src[t];
}

// 13-way classifier + transpose-store into (B, NFRAME, NPTS, NCLS)
__global__ void fc2_kernel(const float* __restrict__ y,      // (B,128,NPTS)
                           const float* __restrict__ W2,     // (13,128)
                           const float* __restrict__ b2,
                           float* __restrict__ out, int frame)
{
    long total = (long)BB * NPTS * NCLS;
    long t = (long)blockIdx.x * blockDim.x + threadIdx.x;
    if (t >= total) return;
    int o = (int)(t % NCLS); long r = t / NCLS;
    int n = (int)(r % NPTS);
    int b = (int)(r / NPTS);
    float v = b2[o];
    for (int c = 0; c < 128; ++c)
        v += W2[o * 128 + c] * y[((long)b * 128 + c) * NPTS + n];
    out[(((long)b * NFRAME + frame) * NPTS + n) * NCLS + o] = v;
}

// ---------------------------------------------------------------------------
// Host orchestration
// ---------------------------------------------------------------------------
struct ConvBN { const float *b, *be, *g, *w; };

static void run_layer(const ConvBN& L, const float* X, float* Y,
                      int M, int K, int S, int G,
                      float* mean, float* inv, hipStream_t stream)
{
    dim3 grid((S + 63) / 64, (M + 15) / 16, G);
    gemm_wmma_kernel<<<grid, 128, 0, stream>>>(L.w, X, Y, L.b, M, K, S, G,
                                               (long)K * S, (long)S, 1L);
    bn_stats_kernel<<<M, 256, 0, stream>>>(Y, mean, inv, M, S, G, 1e-5f);
    long tot = (long)G * M * S;
    bn_relu_kernel<<<(int)((tot + 255) / 256), 256, 0, stream>>>(Y, mean, inv,
                                                                 L.g, L.be, M, S, tot);
}

static void sa_scale(const float* xyz, const float* nx, const float* feats,
                     int Npts, int P, int Cf, float radius, int ns,
                     const ConvBN* L, const int* C,
                     float* A, float* Bf, int* bqIdx,
                     float* featOut, int Ctot, int cOff,
                     float* mean, float* inv, hipStream_t stream)
{
    ball_query_kernel<<<(BB * P + 255) / 256, 256, 0, stream>>>(
        xyz, nx, Npts, P, radius * radius, ns, bqIdx);
    int Cg = 3 + Cf;
    long tg = (long)BB * Cg * P * ns;
    group_kernel<<<(int)((tg + 255) / 256), 256, 0, stream>>>(
        xyz, nx, feats, bqIdx, Npts, P, ns, Cf, A);
    int S = P * ns;
    float* a = A; float* b = Bf;
    for (int l = 0; l < 3; ++l) {
        run_layer(L[l], a, b, C[l + 1], C[l], S, BB, mean, inv, stream);
        float* t = a; a = b; b = t;
    }
    long tp = (long)BB * C[3] * P;
    maxpool_kernel<<<(int)((tp + 255) / 256), 256, 0, stream>>>(
        a, featOut, C[3], P, ns, Ctot, cOff);
}

static void fp_level(const float* known, int Ck, int Nk,
                     const int* idx, const float* wgt,
                     const float* skip, int Cs, int Nu,
                     const ConvBN& L0, const ConvBN& L1, int C1, int C2,
                     float* bufIn, float* bufMid, float* bufOut,
                     float* mean, float* inv, hipStream_t stream)
{
    int Cin = Ck + Cs;
    long tot = (long)BB * Cin * Nu;
    interp_concat_kernel<<<(int)((tot + 255) / 256), 256, 0, stream>>>(
        known, idx, wgt, skip, Ck, Cs, Nu, Nk, bufIn);
    run_layer(L0, bufIn, bufMid, C1, Cin, Nu, BB, mean, inv, stream);
    run_layer(L1, bufMid, bufOut, C2, C1, Nu, BB, mean, inv, stream);
}

extern "C" void kernel_launch(void* const* d_in, const int* in_sizes, int n_in,
                              void* d_out, int out_size, void* d_ws, size_t ws_size,
                              hipStream_t stream)
{
    auto PF = [&](int i) { return (const float*)d_in[i]; };
    auto CB = [&](int base) {
        ConvBN c{PF(base + 0), PF(base + 1), PF(base + 2), PF(base + 3)};
        return c;
    };

    // ---- parameter map (tree leaves: dicts sorted by key, lists in order) ----
    const float* pc  = PF(0);
    const float* h0  = PF(1);
    const float* c0  = PF(2);
    const float* fc_b2 = PF(3);
    ConvBN fc_c1 = CB(4);
    const float* fc_w2 = PF(8);
    ConvBN fp0[2] = { CB(9),  CB(13) };
    ConvBN fp1[2] = { CB(17), CB(21) };
    ConvBN fp2[2] = { CB(25), CB(29) };
    ConvBN fp3[2] = { CB(33), CB(37) };
    const float* b_hh = PF(41);
    const float* b_ih = PF(42);
    const float* w_hh = PF(43);
    const float* w_ih = PF(44);
    ConvBN sa0s0[3] = { CB(45),  CB(49),  CB(53)  };
    ConvBN sa0s1[3] = { CB(57),  CB(61),  CB(65)  };
    ConvBN sa1s0[3] = { CB(69),  CB(73),  CB(77)  };
    ConvBN sa1s1[3] = { CB(81),  CB(85),  CB(89)  };
    ConvBN sa2s0[3] = { CB(93),  CB(97),  CB(101) };
    ConvBN sa2s1[3] = { CB(105), CB(109), CB(113) };
    ConvBN sa3[3]   = { CB(117), CB(121), CB(125) };

    static const int C_SA1_0[4] = {9, 16, 16, 32};
    static const int C_SA1_1[4] = {9, 32, 32, 64};
    static const int C_SA2_0[4] = {99, 64, 64, 128};
    static const int C_SA2_1[4] = {99, 64, 96, 128};
    static const int C_SA3[4]   = {259, 128, 196, 256};
    static const int C_SA4[4]   = {515, 256, 512, 1024};

    // ---- workspace bump allocator ----
    char* wsp = (char*)d_ws;
    auto allocF = [&](size_t n) {
        float* p = (float*)wsp;
        wsp += ((n * sizeof(float)) + 255) & ~(size_t)255;
        return p;
    };
    auto allocI = [&](size_t n) {
        int* p = (int*)wsp;
        wsp += ((n * sizeof(int)) + 255) & ~(size_t)255;
        return p;
    };

    float* xyz0  = allocF((size_t)BB * NPTS * 3);
    float* feats0= allocF((size_t)BB * 6 * NPTS);
    float* xyz1  = allocF((size_t)BB * 1024 * 3);
    float* xyz2  = allocF((size_t)BB * 256 * 3);
    float* xyz3  = allocF((size_t)BB * 64 * 3);
    int*   fpsIdx= allocI((size_t)BB * 1024);
    int*   bqIdx = allocI((size_t)BB * 1024 * 32);
    float* feat1 = allocF((size_t)BB * 96 * 1024);
    float* feat2 = allocF((size_t)BB * 256 * 256);
    float* feat3 = allocF((size_t)BB * 512 * 64);
    float* gf    = allocF((size_t)BB * 1024);
    int*   idx0  = allocI((size_t)BB * 1024 * 3);
    float* w0    = allocF((size_t)BB * 1024 * 3);
    int*   idx1  = allocI((size_t)BB * 1024 * 3);
    float* w1    = allocF((size_t)BB * 1024 * 3);
    int*   idx2  = allocI((size_t)BB * 256 * 3);
    float* w2    = allocF((size_t)BB * 256 * 3);
    int*   idx3  = allocI((size_t)BB * 64 * 3);
    float* w3    = allocF((size_t)BB * 64 * 3);
    float* hbuf  = allocF((size_t)BB * HDIM);
    float* cbuf  = allocF((size_t)BB * HDIM);
    float* gates1= allocF((size_t)4 * HDIM * BB);
    float* gates2= allocF((size_t)4 * HDIM * BB);
    float* meanB = allocF(4096);
    float* invB  = allocF(4096);
    float* P0    = allocF((size_t)10000000);     // max FP input (B,608,1024)
    float* P1    = allocF((size_t)10000000);
    float* G0    = allocF((size_t)33554432);     // max (B,64,1024*32)
    float* G1    = allocF((size_t)33554432);

    // ---- stage 0: split pointcloud ----
    prep_kernel<<<(BB * NPTS + 255) / 256, 256, 0, stream>>>(pc, xyz0, feats0, 6);

    // ---- SA1 ----
    fps_kernel<<<BB, 256, 0, stream>>>(xyz0, 1024, 1024, xyz1, fpsIdx);
    sa_scale(xyz0, xyz1, feats0, 1024, 1024, 6, 0.05f, 16, sa0s0, C_SA1_0,
             G0, G1, bqIdx, feat1, 96, 0,  meanB, invB, stream);
    sa_scale(xyz0, xyz1, feats0, 1024, 1024, 6, 0.10f, 32, sa0s1, C_SA1_1,
             G0, G1, bqIdx, feat1, 96, 32, meanB, invB, stream);

    // ---- SA2 ----
    fps_kernel<<<BB, 256, 0, stream>>>(xyz1, 1024, 256, xyz2, fpsIdx);
    sa_scale(xyz1, xyz2, feat1, 1024, 256, 96, 0.10f, 16, sa1s0, C_SA2_0,
             G0, G1, bqIdx, feat2, 256, 0,   meanB, invB, stream);
    sa_scale(xyz1, xyz2, feat1, 1024, 256, 96, 0.20f, 32, sa1s1, C_SA2_1,
             G0, G1, bqIdx, feat2, 256, 128, meanB, invB, stream);

    // ---- SA3 ----
    fps_kernel<<<BB, 256, 0, stream>>>(xyz2, 256, 64, xyz3, fpsIdx);
    sa_scale(xyz2, xyz3, feat2, 256, 64, 256, 0.20f, 16, sa2s0, C_SA3,
             G0, G1, bqIdx, feat3, 512, 0,   meanB, invB, stream);
    sa_scale(xyz2, xyz3, feat2, 256, 64, 256, 0.40f, 32, sa2s1, C_SA3,
             G0, G1, bqIdx, feat3, 512, 256, meanB, invB, stream);

    // ---- SA4 (group-all -> global feature) ----
    {
        long tg = (long)BB * 515 * 64;
        group_all_kernel<<<(int)((tg + 255) / 256), 256, 0, stream>>>(
            xyz3, feat3, 64, 512, G0);
        float* a = G0; float* b = G1;
        for (int l = 0; l < 3; ++l) {
            run_layer(sa3[l], a, b, C_SA4[l + 1], C_SA4[l], 64, BB, meanB, invB, stream);
            float* t = a; a = b; b = t;
        }
        long tp = (long)BB * 1024;
        maxpool_kernel<<<(int)((tp + 255) / 256), 256, 0, stream>>>(
            a, gf, 1024, 1, 64, 1024, 0);
    }

    // ---- three-NN caches ----
    three_nn_kernel<<<(BB * 1024 + 255) / 256, 256, 0, stream>>>(xyz0, xyz1, 1024, 1024, idx0, w0);
    three_nn_kernel<<<(BB * 1024 + 255) / 256, 256, 0, stream>>>(xyz1, xyz2, 1024, 256,  idx1, w1);
    three_nn_kernel<<<(BB * 256  + 255) / 256, 256, 0, stream>>>(xyz2, xyz3, 256,  64,   idx2, w2);
    nn_const_kernel<<<(BB * 64 + 255) / 256, 256, 0, stream>>>(idx3, w3, BB * 64);

    // ---- LSTM state init ----
    copy_kernel<<<(BB * HDIM + 255) / 256, 256, 0, stream>>>(h0, hbuf, (long)BB * HDIM);
    copy_kernel<<<(BB * HDIM + 255) / 256, 256, 0, stream>>>(c0, cbuf, (long)BB * HDIM);

    // ---- frames ----
    for (int t = 0; t < NFRAME; ++t) {
        const float* xin = (t == 0) ? gf : hbuf;
        dim3 ggrid((BB + 63) / 64, (4 * HDIM + 15) / 16, 1);
        // gates: X[k,n] = x[n*H + k]  ->  sB=0, sK=1, sS=H
        gemm_wmma_kernel<<<ggrid, 128, 0, stream>>>(w_ih, xin,  gates1, b_ih,
                                                    4 * HDIM, HDIM, BB, 1, 0L, 1L, (long)HDIM);
        gemm_wmma_kernel<<<ggrid, 128, 0, stream>>>(w_hh, hbuf, gates2, b_hh,
                                                    4 * HDIM, HDIM, BB, 1, 0L, 1L, (long)HDIM);
        lstm_cell_kernel<<<(BB * HDIM + 255) / 256, 256, 0, stream>>>(gates1, gates2, hbuf, cbuf);

        // FP chain: lvl3 -> lvl2 -> lvl1 -> lvl0
        fp_level(hbuf, 1024, 1,   idx3, w3, feat3, 512, 64,
                 fp3[0], fp3[1], 512, 512, P0, P1, G0, meanB, invB, stream);
        fp_level(G0,   512, 64,   idx2, w2, feat2, 256, 256,
                 fp2[0], fp2[1], 512, 512, P0, P1, G1, meanB, invB, stream);
        fp_level(G1,   512, 256,  idx1, w1, feat1, 96,  1024,
                 fp1[0], fp1[1], 256, 256, P0, P1, G0, meanB, invB, stream);
        fp_level(G0,   256, 1024, idx0, w0, feats0, 6,  1024,
                 fp0[0], fp0[1], 128, 128, P0, P1, G1, meanB, invB, stream);

        // head
        run_layer(fc_c1, G1, P0, 128, 128, 1024, BB, meanB, invB, stream);
        long tf = (long)BB * NPTS * NCLS;
        fc2_kernel<<<(int)((tf + 255) / 256), 256, 0, stream>>>(
            P0, fc_w2, fc_b2, (float*)d_out, t);
    }
}
